// WaveformDecoder_13580686590630
// MI455X (gfx1250) — matrix operand order repaired and checked by reading, not compiled, not run on hardware
//
#include <hip/hip_runtime.h>
#include <hip/hip_bf16.h>

// ---------------------------------------------------------------------------
// WaveNet-style decoder for MI455X (gfx1250, wave32, WMMA 16x16x32 bf16,
// async global->LDS staging of the dilated-conv taps).
// ---------------------------------------------------------------------------

#define BATCH 2
#define CIN   256
#define TIN   128
#define UPK   256
#define CHN   64        // residual channels
#define LQ    32768     // stream length = TIN * UPK
#define NBLK  30
#define LT    64        // L-tile per workgroup in block kernel

typedef __attribute__((ext_vector_type(16))) __bf16 bvec16;
typedef __attribute__((ext_vector_type(8)))  float  fvec8;

__device__ __forceinline__ unsigned short f2bf(float f) {
  unsigned int u = __float_as_uint(f);
  unsigned int r = (u + 0x7FFFu + ((u >> 16) & 1u)) >> 16;  // RNE
  return (unsigned short)r;
}

// A-fragment element -> source-channel mapping (contiguous K order, matches B):
// lane[3:0] = M row, lane[4]=khalf picks K-offset 16; element e -> k = 16*khalf + e.
__device__ __forceinline__ int afrag_k(int e, int khalf) {
  return 16 * khalf + e;
}

// --------------------------- weight staging --------------------------------
// Downcast to bf16 AND pre-swizzle into per-lane WMMA A-fragment order so the
// block kernel loads each fragment as one contiguous 32B vector.
// WgA: [NBLK][8 mtile][3 tap][2 kc][32 lane][16 e]
// WrA: [NBLK][4 mtile][2 kc][32 lane][16 e]
__global__ __launch_bounds__(256) void cvt_kernel(
    const float* __restrict__ Wg, const float* __restrict__ Wr,
    unsigned short* __restrict__ WgA, unsigned short* __restrict__ WrA) {
  int idx = blockIdx.x * 256 + threadIdx.x;
  const int NG = NBLK * 8 * 3 * 2 * 32 * 16;   // 737280
  const int NR = NBLK * 4 * 2 * 32 * 16;       // 122880
  if (idx < NG) {
    int e = idx & 15;
    int tmp = idx >> 4;
    int lane = tmp & 31; tmp >>= 5;
    int kc = tmp & 1;    tmp >>= 1;
    int t = tmp % 3;     tmp /= 3;
    int m = tmp & 7;
    int i = tmp >> 3;
    int khalf = lane >> 4;
    int mrow = 16 * m + (lane & 15);
    int c = 32 * kc + afrag_k(e, khalf);
    WgA[idx] = f2bf(Wg[(((size_t)i * 2 * CHN + mrow) * CHN + c) * 3 + t]);
  }
  if (idx < NR) {
    int e = idx & 15;
    int tmp = idx >> 4;
    int lane = tmp & 31; tmp >>= 5;
    int kc = tmp & 1;    tmp >>= 1;
    int m1 = tmp & 3;
    int i = tmp >> 2;
    int khalf = lane >> 4;
    int mrow = 16 * m1 + (lane & 15);
    int c = 32 * kc + afrag_k(e, khalf);
    WrA[idx] = f2bf(Wr[((size_t)i * CHN + mrow) * CHN + c]);
  }
}

// --------------------------- upsample (block-wise ConvTranspose) -----------
// y[b][o][t*UPK+k] = b_up[o] + sum_c x[b][c][t] * Wup[c][o][k]
// Also writes the bf16 stream copy ybf[b][l][c] ([l][c] layout for async).
__global__ __launch_bounds__(256) void upsample_kernel(
    const float* __restrict__ x, const float* __restrict__ Wup,
    const float* __restrict__ bup, float* __restrict__ y,
    unsigned short* __restrict__ ybf) {
  int t = blockIdx.x;     // 0..TIN-1
  int o = blockIdx.y;     // 0..CHN-1
  int b = blockIdx.z;     // 0..BATCH-1
  int k = threadIdx.x;    // 0..UPK-1
  float acc = bup[o];
  #pragma unroll 4
  for (int c = 0; c < CIN; ++c) {
    float xv = x[((size_t)b * CIN + c) * TIN + t];
    acc = fmaf(xv, Wup[((size_t)c * CHN + o) * UPK + k], acc);
  }
  size_t l = (size_t)t * UPK + k;
  y[((size_t)b * CHN + o) * LQ + l] = acc;
  ybf[((size_t)b * LQ + l) * CHN + o] = f2bf(acc);
}

// --------------------------- gated residual block (WMMA) -------------------
// z[128,L] = dilated_conv(y, Wg) + bg ; g = tanh(z1)*sigmoid(z2)
// yout = yin + Wr @ g + br   (fp32 stream + bf16 [l][c] copy, ping-pong)
__global__ __launch_bounds__(256) void block_kernel(
    const float* __restrict__ yin, float* __restrict__ yout,
    const unsigned short* __restrict__ ybf_in,   // bf16 [B][L][CHN]
    unsigned short* __restrict__ ybf_out,
    const unsigned short* __restrict__ WgA,      // swizzled A frags (this blk)
    const float* __restrict__ bg,                // [128]
    const unsigned short* __restrict__ WrA,      // swizzled A frags (this blk)
    const float* __restrict__ br,                // [64]
    int d) {
  const int tileL = blockIdx.x * LT;
  const int b     = blockIdx.y;
  const float* __restrict__ Yb = yin  + (size_t)b * CHN * LQ;
  float* __restrict__       Yo = yout + (size_t)b * CHN * LQ;
  const unsigned short* __restrict__ YbfI = ybf_in  + (size_t)b * LQ * CHN;
  unsigned short* __restrict__       YbfO = ybf_out + (size_t)b * LQ * CHN;

  __shared__ __align__(32) unsigned short ytap[3][LT][CHN];   // bf16, [l][c]
  __shared__ __align__(32) float          zact[2 * CHN][LT];  // activated z
  __shared__ __align__(32) unsigned short G[LT][CHN];         // bf16 gated

  const int tid   = threadIdx.x;
  const int wave  = tid >> 5;
  const int lane  = tid & 31;
  const int n16   = lane & 15;   // N column (B/C/D layouts)
  const int khalf = lane >> 4;   // lane-half (K/M split)

  // prefetch the fp32 residual tile rows we re-read at the end
  if (tid < CHN) __builtin_prefetch(&Yb[(size_t)tid * LQ + tileL], 0, 0);

  // ---- async-stage the three dilated taps (bf16, [l][c]) into LDS ----
  // 16B chunks: (tap, l, 8-channel segment). OOB (conv padding) -> zeros.
  for (int pos = tid; pos < 3 * LT * 8; pos += 256) {
    int tap = pos / (LT * 8);
    int rem = pos % (LT * 8);
    int ll  = rem >> 3;
    int seg = rem & 7;
    int gl  = tileL + ll + (tap - 1) * d;
    unsigned short* lp = &ytap[tap][ll][seg * 8];
    if (gl >= 0 && gl < LQ) {
      const unsigned short* gp = &YbfI[(size_t)gl * CHN + seg * 8];
      unsigned lds_off = (unsigned)(size_t)lp;   // low 32 bits = LDS address
      asm volatile("global_load_async_to_lds_b128 %0, %1, off"
                   :: "v"(lds_off), "v"(gp) : "memory");
    } else {
      uint4 zz{};
      *(uint4*)lp = zz;
    }
  }
  asm volatile("s_wait_asynccnt 0x0" ::: "memory");
  __syncthreads();

  // ---- A fragments for the gated conv (contiguous 32B per lane) ----
  bvec16 Ag[3][2];
  #pragma unroll
  for (int t = 0; t < 3; ++t)
    #pragma unroll
    for (int kc = 0; kc < 2; ++kc)
      Ag[t][kc] = *(const bvec16*)
          &WgA[((((size_t)wave * 3 + t) * 2 + kc) * 32 + lane) * 16];

  // ---- accumulators seeded with conv bias (C/D f32 layout) ----
  fvec8 bias0;
  #pragma unroll
  for (int r = 0; r < 8; ++r) bias0[r] = bg[16 * wave + r + 8 * khalf];
  fvec8 acc[4];
  #pragma unroll
  for (int n = 0; n < 4; ++n) acc[n] = bias0;

  // ---- z = sum_taps Wg_tap @ ytap : 3 taps x K=64 x 4 N-tiles ----
  #pragma unroll
  for (int t = 0; t < 3; ++t) {
    #pragma unroll
    for (int kc = 0; kc < 2; ++kc) {
      #pragma unroll
      for (int n = 0; n < 4; ++n) {
        const unsigned short* src = &ytap[t][n * 16 + n16][32 * kc + 16 * khalf];
        bvec16 bf = *(const bvec16*)src;
        acc[n] = __builtin_amdgcn_wmma_f32_16x16x32_bf16(
            false, Ag[t][kc], false, bf, (short)0, acc[n], false, false);
      }
    }
  }

  // ---- apply tanh (rows 0..63) / sigmoid (rows 64..127) via fast exp ----
  const bool isTanh = (wave < 4);
  #pragma unroll
  for (int n = 0; n < 4; ++n) {
    #pragma unroll
    for (int r = 0; r < 8; ++r) {
      int row = 16 * wave + r + 8 * khalf;
      int col = n * 16 + n16;
      float v = acc[n][r];
      float a;
      if (isTanh) {
        float e2 = __expf(2.0f * v);
        a = 1.0f - 2.0f * __builtin_amdgcn_rcpf(e2 + 1.0f);
      } else {
        a = __builtin_amdgcn_rcpf(1.0f + __expf(-v));
      }
      zact[row][col] = a;
    }
  }
  __syncthreads();

  // ---- gating: G[l][c] = bf16(tanh(z1) * sigmoid(z2)) ----
  for (int pos = tid; pos < CHN * LT; pos += 256) {
    int c  = pos / LT;
    int ll = pos % LT;
    G[ll][c] = f2bf(zact[c][ll] * zact[c + CHN][ll]);
  }
  __syncthreads();

  // ---- 1x1 residual GEMM: res = Wr @ G + br ; yout = yin + res ----
  const int m1 = wave & 3;           // waves 0-3 / 4-7 split the 4 N tiles
  bvec16 Ar[2];
  #pragma unroll
  for (int kc = 0; kc < 2; ++kc)
    Ar[kc] = *(const bvec16*)
        &WrA[(((size_t)m1 * 2 + kc) * 32 + lane) * 16];

  #pragma unroll
  for (int nn = 0; nn < 2; ++nn) {
    int n = (wave >> 2) * 2 + nn;
    fvec8 c0;
    #pragma unroll
    for (int r = 0; r < 8; ++r) c0[r] = br[16 * m1 + r + 8 * khalf];
    #pragma unroll
    for (int kc = 0; kc < 2; ++kc) {
      const unsigned short* src = &G[n * 16 + n16][32 * kc + 16 * khalf];
      bvec16 bf = *(const bvec16*)src;
      c0 = __builtin_amdgcn_wmma_f32_16x16x32_bf16(
          false, Ar[kc], false, bf, (short)0, c0, false, false);
    }
    int col = tileL + n * 16 + n16;
    union { uint4 q; unsigned short s[8]; } pk;
    #pragma unroll
    for (int r = 0; r < 8; ++r) {
      int row = 16 * m1 + r + 8 * khalf;     // 8 consecutive channels
      float v = Yb[(size_t)row * LQ + col] + c0[r];
      Yo[(size_t)row * LQ + col] = v;
      pk.s[r] = f2bf(v);
    }
    // one 16B store: 8 consecutive bf16 channels of ybf[l][c]
    *(uint4*)&YbfO[(size_t)col * CHN + 16 * m1 + 8 * khalf] = pk.q;
  }
}

// --------------------------- final 1x64 projection -------------------------
__global__ __launch_bounds__(256) void final_kernel(
    const float* __restrict__ y, const float* __restrict__ Wf,
    const float* __restrict__ bf_, float* __restrict__ out) {
  int idx = blockIdx.x * 256 + threadIdx.x;     // over BATCH*LQ
  if (idx >= BATCH * LQ) return;
  int b = idx / LQ, l = idx % LQ;
  const float* Yb = y + (size_t)b * CHN * LQ;
  float acc = bf_[0];
  #pragma unroll
  for (int c = 0; c < CHN; ++c) acc = fmaf(Wf[c], Yb[(size_t)c * LQ + l], acc);
  out[idx] = acc;
}

// ---------------------------------------------------------------------------
extern "C" void kernel_launch(void* const* d_in, const int* in_sizes, int n_in,
                              void* d_out, int out_size, void* d_ws, size_t ws_size,
                              hipStream_t stream) {
  const float* x    = (const float*)d_in[0];   // [2,256,128]
  const float* Wup  = (const float*)d_in[1];   // [256,64,256]
  const float* bup  = (const float*)d_in[2];   // [64]
  const float* Wg   = (const float*)d_in[3];   // [30,128,64,3]
  const float* bg   = (const float*)d_in[4];   // [30,128]
  const float* Wr   = (const float*)d_in[5];   // [30,64,64]
  const float* br   = (const float*)d_in[6];   // [30,64]
  const float* Wf   = (const float*)d_in[7];   // [1,64]
  const float* bf_  = (const float*)d_in[8];   // [1]
  float* out = (float*)d_out;                  // [2,32768]

  // workspace layout
  char* ws = (char*)d_ws;
  const size_t ybytes  = (size_t)BATCH * CHN * LQ * sizeof(float);   // 16 MB
  const size_t ybbytes = (size_t)BATCH * LQ * CHN * 2;               //  8 MB
  float* y0 = (float*)(ws);
  float* y1 = (float*)(ws + ybytes);
  unsigned short* ybf0 = (unsigned short*)(ws + 2 * ybytes);
  unsigned short* ybf1 = (unsigned short*)(ws + 2 * ybytes + ybbytes);
  unsigned short* WgA  = (unsigned short*)(ws + 2 * ybytes + 2 * ybbytes);
  unsigned short* WrA  = (unsigned short*)(ws + 2 * ybytes + 2 * ybbytes +
                                           (size_t)NBLK * 2 * CHN * CHN * 3 * 2);

  // 1) bf16 weight staging + WMMA A-fragment swizzle
  {
    int n = NBLK * 2 * CHN * CHN * 3;
    cvt_kernel<<<(n + 255) / 256, 256, 0, stream>>>(Wg, Wr, WgA, WrA);
  }

  // 2) upsample into y0 (+ bf16 copy ybf0)
  upsample_kernel<<<dim3(TIN, CHN, BATCH), 256, 0, stream>>>(
      x, Wup, bup, y0, ybf0);

  // 3) 30 gated residual blocks, ping-pong (y0,ybf0) <-> (y1,ybf1)
  dim3 bgrid(LQ / LT, BATCH);
  for (int i = 0; i < NBLK; ++i) {
    int dil = 1 << (i % 10);
    const float*          yi = (i & 1) ? y1 : y0;
    float*                yo = (i & 1) ? y0 : y1;
    const unsigned short* bi = (i & 1) ? ybf1 : ybf0;
    unsigned short*       bo = (i & 1) ? ybf0 : ybf1;
    block_kernel<<<bgrid, 256, 0, stream>>>(
        yi, yo, bi, bo,
        WgA + (size_t)i * 8 * 3 * 2 * 32 * 16,
        bg  + (size_t)i * 2 * CHN,
        WrA + (size_t)i * 4 * 2 * 32 * 16,
        br  + (size_t)i * CHN,
        dil);
  }
  // NBLK = 30 (even) -> final fp32 stream is in y0

  // 4) final projection
  final_kernel<<<(BATCH * LQ + 255) / 256, 256, 0, stream>>>(y0, Wf, bf_, out);
}
